// SheafNN_4466765988557
// MI455X (gfx1250) — matrix operand order repaired and checked by reading, not compiled, not run on hardware
//
#include <hip/hip_runtime.h>
#include <hip/hip_bf16.h>
#include <stdint.h>

typedef __attribute__((ext_vector_type(16))) __bf16 v16bf;
typedef __attribute__((ext_vector_type(8)))  __bf16 v8bf;
typedef __attribute__((ext_vector_type(8)))  float  v8f;
typedef __attribute__((ext_vector_type(4)))  float  v4f;
typedef __attribute__((ext_vector_type(2)))  float  v2f;

#define NN   20000
#define EE   200000
#define IND  500
#define HIDD 256
#define OUTD 16

#define EW   8   // waves per block in edge-MLP

#if defined(__AMDGCN__) && __has_builtin(__builtin_amdgcn_tensor_load_to_lds) && __has_builtin(__builtin_amdgcn_s_wait_tensorcnt)
#define HAVE_TDM 1
typedef __attribute__((ext_vector_type(4))) unsigned int u32x4;
typedef __attribute__((ext_vector_type(8))) int          i32x8;
typedef __attribute__((ext_vector_type(4))) int          i32x4;
#endif

// ---------------------------------------------------------------------------
// Fragment loaders (contiguous k order shared by A and B), vectorized as b128 chunks.
// A (16x32 bf16): lane(m,half) holds K = [16h .. 16h+15] contiguous.
// B (32x16 bf16): lane(n,half) holds K = [16h .. 16h+15] contiguous.
// ---------------------------------------------------------------------------
__device__ __forceinline__ v16bf ld_afrag(const __bf16* base_k0, int half) {
    const v8bf lo = *(const v8bf*)(base_k0 + 16 * half);
    const v8bf hi = *(const v8bf*)(base_k0 + 16 * half + 8);
    v16bf a;
#pragma unroll
    for (int i = 0; i < 8; ++i) { a[i] = lo[i]; a[8 + i] = hi[i]; }
    return a;
}
__device__ __forceinline__ v16bf ld_bfrag(const __bf16* p) {
    const v8bf lo = *(const v8bf*)p;
    const v8bf hi = *(const v8bf*)(p + 8);
    v16bf b;
#pragma unroll
    for (int i = 0; i < 8; ++i) { b[i] = lo[i]; b[8 + i] = hi[i]; }
    return b;
}

// ---------------------------------------------------------------------------
// Conversion / utility kernels
// ---------------------------------------------------------------------------
__global__ void k_cvt(const float* __restrict__ s, __bf16* __restrict__ d, int n) {
    int i = blockIdx.x * blockDim.x + threadIdx.x;
    if (i < n) d[i] = (__bf16)s[i];
}
__global__ void k_cvt_pad(const float* __restrict__ s, __bf16* __restrict__ d,
                          int rows, int cols, int colsp) {
    int i = blockIdx.x * blockDim.x + threadIdx.x;
    if (i < rows * colsp) {
        int r = i / colsp, c = i - r * colsp;
        d[i] = (c < cols) ? (__bf16)s[(size_t)r * cols + c] : (__bf16)0.0f;
    }
}
__global__ void k_zero(float* __restrict__ p, int n) {
    int i = blockIdx.x * blockDim.x + threadIdx.x;
    if (i < n) p[i] = 0.0f;
}

// ---------------------------------------------------------------------------
// Input MLP: h = relu(x @ Win1^T + bin1) @ Win2^T + bin2. One wave / 16 nodes.
// x pre-padded to bf16 [N][512].
// ---------------------------------------------------------------------------
__global__ void __launch_bounds__(32) k_mlp_in(
    const __bf16* __restrict__ xbf, const __bf16* __restrict__ Win1p,
    const float* __restrict__ bin1, const __bf16* __restrict__ Win2b,
    const float* __restrict__ bin2, float* __restrict__ hout,
    __bf16* __restrict__ hbf)
{
    __shared__ __bf16 t[16 * 64];
    const int lane = threadIdx.x;
    const int m = lane & 15, half = lane >> 4;
    const int tile = blockIdx.x;
    const __bf16* xrow = xbf + (size_t)(tile * 16 + m) * 512;

    v8f acc[4];
#pragma unroll
    for (int c = 0; c < 4; ++c)
#pragma unroll
        for (int i = 0; i < 8; ++i) acc[c][i] = 0.0f;

    for (int k0 = 0; k0 < 512; k0 += 32) {
        v16bf a = ld_afrag(xrow + k0, half);
#pragma unroll
        for (int c = 0; c < 4; ++c) {
            v16bf b = ld_bfrag(Win1p + (size_t)(c * 16 + m) * 512 + k0 + 16 * half);
            acc[c] = __builtin_amdgcn_wmma_f32_16x16x32_bf16(
                false, a, false, b, (short)0, acc[c], false, false);
        }
    }
#pragma unroll
    for (int c = 0; c < 4; ++c) {
        int n = c * 16 + m;
        float bb = bin1[n];
#pragma unroll
        for (int r = 0; r < 8; ++r) {
            int mm = r + 8 * half;
            float v = acc[c][r] + bb;
            t[mm * 64 + n] = (__bf16)(v > 0.0f ? v : 0.0f);
        }
    }
    __syncthreads();

    v16bf a2[2];
#pragma unroll
    for (int ks = 0; ks < 2; ++ks) a2[ks] = ld_afrag(t + m * 64 + ks * 32, half);

    for (int c = 0; c < 16; ++c) {
        v8f acc2;
#pragma unroll
        for (int i = 0; i < 8; ++i) acc2[i] = 0.0f;
#pragma unroll
        for (int ks = 0; ks < 2; ++ks) {
            v16bf b = ld_bfrag(Win2b + (size_t)(c * 16 + m) * 64 + ks * 32 + 16 * half);
            acc2 = __builtin_amdgcn_wmma_f32_16x16x32_bf16(
                false, a2[ks], false, b, (short)0, acc2, false, false);
        }
        int n = c * 16 + m;
        float bb = bin2[n];
#pragma unroll
        for (int r = 0; r < 8; ++r) {
            int mm = r + 8 * half;
            float v = acc2[r] + bb;
            size_t idx = (size_t)(tile * 16 + mm) * HIDD + n;
            hout[idx] = v;
            hbf[idx]  = (__bf16)v;
        }
    }
}

// ---------------------------------------------------------------------------
// Edge MLP. 8 waves / block; Wm1 (64KB bf16) staged to LDS via TDM once per
// block; gathered hs/hd rows read as b128 bf16.
// ---------------------------------------------------------------------------
__global__ void __launch_bounds__(EW * 32) k_edge_mlp(
    const int* __restrict__ src, const int* __restrict__ dst,
    const __bf16* __restrict__ hbf,
    const __bf16* __restrict__ Wm1b, const float* __restrict__ bm1,
    const __bf16* __restrict__ Wm2b, const float* __restrict__ bm2,
    float* __restrict__ Fsb, float* __restrict__ Ftb)
{
    extern __shared__ __bf16 smem[];
    __bf16* Wlds = smem;                               // 64*512
    const int tid = threadIdx.x;
    const int wid = tid >> 5, lane = tid & 31;
    const int m = lane & 15, half = lane >> 4;
    __bf16* p1 = smem + 64 * 512 + wid * 1024;
    __bf16* p2 = smem + 64 * 512 + EW * 1024 + wid * 1024;

#ifdef HAVE_TDM
    if (wid == 0) {
        // D# per ISA 8.3/8.4: one descriptor, data_size=2B, 1 row of 32768
        // elements, tile == tensor, dest LDS offset 0 (dynamic region base).
        unsigned long long ga = (unsigned long long)(uintptr_t)Wm1b;
        u32x4 g0;
        g0[0] = 1u;                                        // count=1
        g0[1] = 0u;                                        // lds_addr
        g0[2] = (unsigned)(ga & 0xffffffffull);            // global_addr lo
        g0[3] = (unsigned)((ga >> 32) & 0x01ffffffull) | (2u << 30); // type=2
        i32x8 g1;
        g1[0] = (int)(1u << 16);       // data_size=1 (2 bytes), wg_mask=0
        g1[1] = (int)0x80000000u;      // tensor_dim0[15:0]=32768 -> bits[63:48]
        g1[2] = (int)(1u << 16);       // tensor_dim0[31:16]=0, tensor_dim1=1
        g1[3] = (int)0x80000000u;      // tile_dim0=32768 -> bits[127:112]
        g1[4] = 1;                     // tile_dim1=1, tile_dim2=0
        g1[5] = 32768;                 // tensor_dim0_stride lo
        g1[6] = 0;
        g1[7] = 0;
        i32x4 g2; g2[0] = 0; g2[1] = 0; g2[2] = 0; g2[3] = 0;
        i32x4 g3 = g2;
        i32x8 g4;                      // unused trailing SGPR group (clang-23 form)
#pragma unroll
        for (int q = 0; q < 8; ++q) g4[q] = 0;
        __builtin_amdgcn_tensor_load_to_lds(g0, g1, g2, g3, g4, 0);
        __builtin_amdgcn_s_wait_tensorcnt(0);
    }
#else
    for (int i = tid; i < 64 * 512 / 8; i += EW * 32)
        ((v8bf*)Wlds)[i] = ((const v8bf*)Wm1b)[i];
#endif
    __syncthreads();

    const int tile = blockIdx.x * EW + wid;
    if (tile >= EE / 16) return;
    const int e0 = tile * 16;
    const int si = src[e0 + m], di = dst[e0 + m];
    const __bf16* hs = hbf + (size_t)si * HIDD;
    const __bf16* hd = hbf + (size_t)di * HIDD;
    __builtin_prefetch(hs, 0, 1);
    __builtin_prefetch(hd, 0, 1);

    v8f acc1[4], acc2[4];
#pragma unroll
    for (int c = 0; c < 4; ++c)
#pragma unroll
        for (int i = 0; i < 8; ++i) { acc1[c][i] = 0.0f; acc2[c][i] = 0.0f; }

    for (int k0 = 0; k0 < 512; k0 += 32) {
        const __bf16* r1 = (k0 < 256) ? hs : hd;
        const __bf16* r2 = (k0 < 256) ? hd : hs;
        const int koff = (k0 < 256) ? k0 : (k0 - 256);
        v16bf a1 = ld_afrag(r1 + koff, half);
        v16bf a2v = ld_afrag(r2 + koff, half);
#pragma unroll
        for (int c = 0; c < 4; ++c) {
            v16bf b = ld_bfrag(Wlds + (size_t)(c * 16 + m) * 512 + k0 + 16 * half);
            acc1[c] = __builtin_amdgcn_wmma_f32_16x16x32_bf16(
                false, a1, false, b, (short)0, acc1[c], false, false);
            acc2[c] = __builtin_amdgcn_wmma_f32_16x16x32_bf16(
                false, a2v, false, b, (short)0, acc2[c], false, false);
        }
    }
#pragma unroll
    for (int c = 0; c < 4; ++c) {
        int n = c * 16 + m;
        float bb = bm1[n];
#pragma unroll
        for (int r = 0; r < 8; ++r) {
            int mm = r + 8 * half;
            float v1 = acc1[c][r] + bb;
            float v2 = acc2[c][r] + bb;
            p1[mm * 64 + n] = (__bf16)(v1 > 0.0f ? v1 : 0.0f);
            p2[mm * 64 + n] = (__bf16)(v2 > 0.0f ? v2 : 0.0f);
        }
    }
    // p1/p2 are per-wave regions: wave-internal DS ordering suffices.
    v8f am1, am2;
#pragma unroll
    for (int i = 0; i < 8; ++i) { am1[i] = 0.0f; am2[i] = 0.0f; }
#pragma unroll
    for (int ks = 0; ks < 2; ++ks) {
        v16bf a1 = ld_afrag(p1 + m * 64 + ks * 32, half);
        v16bf a2v = ld_afrag(p2 + m * 64 + ks * 32, half);
        v16bf b = ld_bfrag(Wm2b + (size_t)m * 64 + ks * 32 + 16 * half);
        am1 = __builtin_amdgcn_wmma_f32_16x16x32_bf16(false, a1, false, b, (short)0, am1, false, false);
        am2 = __builtin_amdgcn_wmma_f32_16x16x32_bf16(false, a2v, false, b, (short)0, am2, false, false);
    }
    float bb = bm2[m];
#pragma unroll
    for (int r = 0; r < 8; ++r) {
        int mm = r + 8 * half;
        Fsb[(size_t)(e0 + mm) * 16 + m] = am1[r] + bb;
        Ftb[(size_t)(e0 + mm) * 16 + m] = am2[r] + bb;
    }
}

// ---------------------------------------------------------------------------
// Diffusion node transform H = Wd1^T @ xb @ Wd2^T  (f32 WMMA 16x16x4)
// ---------------------------------------------------------------------------
__global__ void __launch_bounds__(32) k_node_H(
    const float* __restrict__ xb, const float* __restrict__ Wd1l,
    const float* __restrict__ Wd2l, float* __restrict__ Hb)
{
    __shared__ float tmp[16 * 64];
    const int lane = threadIdx.x;
    const int m = lane & 15, half = lane >> 4;
    const int node0 = blockIdx.x * 4;

    float wd1[16];
#pragma unroll
    for (int i = 0; i < 16; ++i) wd1[i] = Wd1l[i];

    for (int idx = lane; idx < 1024; idx += 32) {
        int r = idx >> 6, f = idx & 63;
        int nl = r >> 2, i = r & 3;
        const float* xr = xb + (size_t)(node0 + nl) * HIDD + f;
        float s = 0.0f;
#pragma unroll
        for (int j = 0; j < 4; ++j) s += wd1[j * 4 + i] * xr[j * 64];
        tmp[r * 64 + f] = s;
    }
    __syncthreads();

#if __has_builtin(__builtin_amdgcn_wmma_f32_16x16x4_f32)
    v8f acc[4];
#pragma unroll
    for (int c = 0; c < 4; ++c)
#pragma unroll
        for (int i = 0; i < 8; ++i) acc[c][i] = 0.0f;
    for (int k0 = 0; k0 < 64; k0 += 4) {
        v2f a = *(const v2f*)(tmp + m * 64 + k0 + 2 * half);
#pragma unroll
        for (int c = 0; c < 4; ++c) {
            v2f b = *(const v2f*)(Wd2l + (size_t)(c * 16 + m) * 64 + k0 + 2 * half);
            acc[c] = __builtin_amdgcn_wmma_f32_16x16x4_f32(
                false, a, false, b, (short)0, acc[c], false, false);
        }
    }
#pragma unroll
    for (int c = 0; c < 4; ++c) {
        int g = c * 16 + m;
#pragma unroll
        for (int r = 0; r < 8; ++r) {
            int mm = r + 8 * half;
            Hb[(size_t)(node0 + (mm >> 2)) * HIDD + (mm & 3) * 64 + g] = acc[c][r];
        }
    }
#else
    for (int idx = lane; idx < 1024; idx += 32) {
        int r = idx >> 6, g = idx & 63;
        float s = 0.0f;
        for (int f = 0; f < 64; ++f) s += tmp[r * 64 + f] * Wd2l[(size_t)g * 64 + f];
        Hb[(size_t)(node0 + (r >> 2)) * HIDD + (r & 3) * 64 + g] = s;
    }
#endif
}

// ---------------------------------------------------------------------------
// Edge diffusion + scatter via f32 atomics.
// ---------------------------------------------------------------------------
__global__ void __launch_bounds__(32) k_edge_diff(
    const int* __restrict__ src, const int* __restrict__ dst,
    const float* __restrict__ Fsb, const float* __restrict__ Ftb,
    const float* __restrict__ Hb, float* __restrict__ LH)
{
    const int e = blockIdx.x;
    const int lane = threadIdx.x;
    const int s = src[e], d = dst[e];
    v4f fsq[4], ftq[4];
#pragma unroll
    for (int q = 0; q < 4; ++q) {
        fsq[q] = *(const v4f*)(Fsb + (size_t)e * 16 + 4 * q);
        ftq[q] = *(const v4f*)(Ftb + (size_t)e * 16 + 4 * q);
    }
    const float* Hs = Hb + (size_t)s * HIDD;
    const float* Hd = Hb + (size_t)d * HIDD;
    const int f0 = lane * 2;
    v2f hsv[4], hdv[4];
#pragma unroll
    for (int j = 0; j < 4; ++j) {
        hsv[j] = *(const v2f*)(Hs + j * 64 + f0);
        hdv[j] = *(const v2f*)(Hd + j * 64 + f0);
    }
    float bx[4][2];
#pragma unroll
    for (int i = 0; i < 4; ++i)
#pragma unroll
        for (int p = 0; p < 2; ++p) {
            float a = 0.0f;
#pragma unroll
            for (int j = 0; j < 4; ++j)
                a += ftq[i][j] * hdv[j][p] - fsq[i][j] * hsv[j][p];
            bx[i][p] = a;
        }
#pragma unroll
    for (int j = 0; j < 4; ++j)
#pragma unroll
        for (int p = 0; p < 2; ++p) {
            float vs = 0.0f, vd = 0.0f;
#pragma unroll
            for (int i = 0; i < 4; ++i) {
                vs += fsq[i][j] * bx[i][p];
                vd += ftq[i][j] * bx[i][p];
            }
            atomicAdd(&LH[(size_t)s * HIDD + j * 64 + f0 + p], -vs);
            atomicAdd(&LH[(size_t)d * HIDD + j * 64 + f0 + p],  vd);
        }
}

__global__ void k_apply(float* __restrict__ xb, const float* __restrict__ LH, int n) {
    int i = blockIdx.x * blockDim.x + threadIdx.x;
    if (i < n) {
        float l = LH[i];
        xb[i] -= (l > 0.0f ? l : 0.0f);
    }
}

// ---------------------------------------------------------------------------
// Output MLP: out = relu(h @ Wo1^T + bo1) @ Wo2^T + bo2 ; h pre-cvt to bf16.
// ---------------------------------------------------------------------------
__global__ void __launch_bounds__(32) k_mlp_out(
    const __bf16* __restrict__ hbf, const __bf16* __restrict__ Wo1b,
    const float* __restrict__ bo1, const __bf16* __restrict__ Wo2b,
    const float* __restrict__ bo2, float* __restrict__ out)
{
    __shared__ __bf16 t[16 * 64];
    const int lane = threadIdx.x;
    const int m = lane & 15, half = lane >> 4;
    const int tile = blockIdx.x;
    const __bf16* hrow = hbf + (size_t)(tile * 16 + m) * HIDD;

    v8f acc[4];
#pragma unroll
    for (int c = 0; c < 4; ++c)
#pragma unroll
        for (int i = 0; i < 8; ++i) acc[c][i] = 0.0f;

    for (int k0 = 0; k0 < 256; k0 += 32) {
        v16bf a = ld_afrag(hrow + k0, half);
#pragma unroll
        for (int c = 0; c < 4; ++c) {
            v16bf b = ld_bfrag(Wo1b + (size_t)(c * 16 + m) * 256 + k0 + 16 * half);
            acc[c] = __builtin_amdgcn_wmma_f32_16x16x32_bf16(
                false, a, false, b, (short)0, acc[c], false, false);
        }
    }
#pragma unroll
    for (int c = 0; c < 4; ++c) {
        int n = c * 16 + m;
        float bb = bo1[n];
#pragma unroll
        for (int r = 0; r < 8; ++r) {
            int mm = r + 8 * half;
            float v = acc[c][r] + bb;
            t[mm * 64 + n] = (__bf16)(v > 0.0f ? v : 0.0f);
        }
    }
    __syncthreads();

    v8f a2;
#pragma unroll
    for (int i = 0; i < 8; ++i) a2[i] = 0.0f;
#pragma unroll
    for (int ks = 0; ks < 2; ++ks) {
        v16bf a = ld_afrag(t + m * 64 + ks * 32, half);
        v16bf b = ld_bfrag(Wo2b + (size_t)m * 64 + ks * 32 + 16 * half);
        a2 = __builtin_amdgcn_wmma_f32_16x16x32_bf16(false, a, false, b, (short)0, a2, false, false);
    }
    float bb = bo2[m];
#pragma unroll
    for (int r = 0; r < 8; ++r) {
        int mm = r + 8 * half;
        out[(size_t)(tile * 16 + mm) * OUTD + m] = a2[r] + bb;
    }
}

// ---------------------------------------------------------------------------
extern "C" void kernel_launch(void* const* d_in, const int* in_sizes, int n_in,
                              void* d_out, int out_size, void* d_ws, size_t ws_size,
                              hipStream_t stream) {
    const float* x    = (const float*)d_in[0];
    const int*   eidx = (const int*)d_in[1];
    const float* Win1 = (const float*)d_in[2];
    const float* bin1 = (const float*)d_in[3];
    const float* Win2 = (const float*)d_in[4];
    const float* bin2 = (const float*)d_in[5];
    const float* Wm1  = (const float*)d_in[6];
    const float* bm1  = (const float*)d_in[7];
    const float* Wm2  = (const float*)d_in[8];
    const float* bm2  = (const float*)d_in[9];
    const float* Wd1  = (const float*)d_in[10];
    const float* Wd2  = (const float*)d_in[11];
    const float* Wo1  = (const float*)d_in[12];
    const float* bo1  = (const float*)d_in[13];
    const float* Wo2  = (const float*)d_in[14];
    const float* bo2  = (const float*)d_in[15];
    const int* src = eidx;
    const int* dst = eidx + EE;
    float* out = (float*)d_out;

    char* wsb = (char*)d_ws;
    size_t off = 0;
    auto carve = [&](size_t bytes) {
        char* p = wsb + off;
        off += (bytes + 255) & ~(size_t)255;
        return p;
    };
    __bf16* x_bf   = (__bf16*)carve((size_t)NN * 512 * 2);
    __bf16* h_bf   = (__bf16*)carve((size_t)NN * HIDD * 2);
    __bf16* Win1p  = (__bf16*)carve(64 * 512 * 2);
    __bf16* Win2b  = (__bf16*)carve(256 * 64 * 2);
    __bf16* Wm1b   = (__bf16*)carve(64 * 512 * 2);
    __bf16* Wm2b   = (__bf16*)carve(16 * 64 * 2);
    __bf16* Wo1b   = (__bf16*)carve(64 * 256 * 2);
    __bf16* Wo2b   = (__bf16*)carve(16 * 64 * 2);
    float*  h_f32  = (float*)carve((size_t)NN * HIDD * 4);
    float*  Fsb    = (float*)carve((size_t)EE * 16 * 4);
    float*  Ftb    = (float*)carve((size_t)EE * 16 * 4);
    float*  Hb     = (float*)carve((size_t)NN * HIDD * 4);
    float*  LH     = (float*)carve((size_t)NN * HIDD * 4);

    // conversions (x zero-padded 500->512; weights straight)
    k_cvt_pad<<<((size_t)NN * 512 + 255) / 256, 256, 0, stream>>>(x, x_bf, NN, IND, 512);
    k_cvt_pad<<<(64 * 512 + 255) / 256, 256, 0, stream>>>(Win1, Win1p, 64, IND, 512);
    k_cvt<<<(256 * 64 + 255) / 256, 256, 0, stream>>>(Win2, Win2b, 256 * 64);
    k_cvt<<<(64 * 512 + 255) / 256, 256, 0, stream>>>(Wm1, Wm1b, 64 * 512);
    k_cvt<<<(16 * 64 + 255) / 256, 256, 0, stream>>>(Wm2, Wm2b, 16 * 64);
    k_cvt<<<(64 * 256 + 255) / 256, 256, 0, stream>>>(Wo1, Wo1b, 64 * 256);
    k_cvt<<<(16 * 64 + 255) / 256, 256, 0, stream>>>(Wo2, Wo2b, 16 * 64);

    // input MLP
    k_mlp_in<<<NN / 16, 32, 0, stream>>>(x_bf, Win1p, bin1, Win2b, bin2, h_f32, h_bf);

    // edge MLP (Wm1 staged to LDS via TDM)
    const int etiles = EE / 16;
    const int eblocks = (etiles + EW - 1) / EW;
    const size_t esmem = (64 * 512 + 2 * EW * 16 * 64) * sizeof(__bf16);
    k_edge_mlp<<<eblocks, EW * 32, esmem, stream>>>(src, dst, h_bf, Wm1b, bm1,
                                                    Wm2b, bm2, Fsb, Ftb);

    // sheaf diffusion
    const int hsz = NN * HIDD;
    for (int l = 0; l < 2; ++l) {
        k_zero<<<(hsz + 255) / 256, 256, 0, stream>>>(LH, hsz);
        k_node_H<<<NN / 4, 32, 0, stream>>>(h_f32, Wd1 + l * 16, Wd2 + l * 64 * 64, Hb);
        k_edge_diff<<<EE, 32, 0, stream>>>(src, dst, Fsb, Ftb, Hb, LH);
        k_apply<<<(hsz + 255) / 256, 256, 0, stream>>>(h_f32, LH, hsz);
    }

    // output MLP (h -> bf16 once, then WMMA)
    k_cvt<<<(hsz + 255) / 256, 256, 0, stream>>>(h_f32, h_bf, hsz);
    k_mlp_out<<<NN / 16, 32, 0, stream>>>(h_bf, Wo1b, bo1, Wo2b, bo2, out);
}
